// DEQDualPathDenoiser_223338299738
// MI455X (gfx1250) — matrix operand order repaired and checked by reading, not compiled, not run on hardware
//
#include <hip/hip_runtime.h>
#include <hip/hip_bf16.h>

// ---------------------------------------------------------------------------
// DEQ dual-path denoiser for MI455X (gfx1250, wave32, WMMA, async-LDS)
// B=4096, K=16384, D=DH=512, GH=32, MAX_ITER=30, TOL=1e-5
// ---------------------------------------------------------------------------

typedef __attribute__((ext_vector_type(16))) __bf16 v16bf;
typedef __attribute__((ext_vector_type(8)))  float  v8f;
typedef __attribute__((__vector_size__(16))) int    v4i;

#define Bsz   4096
#define Kpat  16384
#define Dd    512
#define GHd   32
#define TOLf  1e-5f

#if __has_builtin(__builtin_amdgcn_global_load_async_to_lds_b128) && \
    __has_builtin(__builtin_amdgcn_s_wait_asynccnt)
#define HAS_ASYNC_LDS 1
#else
#define HAS_ASYNC_LDS 0
#endif

#if HAS_ASYNC_LDS
typedef __attribute__((address_space(1))) v4i* gv4i_ptr;  // global v4i*
typedef __attribute__((address_space(3))) v4i* lv4i_ptr;  // LDS v4i*
#endif

static __device__ __forceinline__ v8f wmma_bf16(v16bf a, v16bf b, v8f c) {
  return __builtin_amdgcn_wmma_f32_16x16x32_bf16(false, a, false, b, (short)0, c,
                                                 false, false);
}

// Issue the 64x512 bf16 V slice [kt, kt+64) into LDS.
// Async path: 32 GLOBAL_LOAD_ASYNC_TO_LDS_B128 per thread, tracked by ASYNCcnt.
static __device__ __forceinline__ void v_tile_issue(const __bf16* __restrict__ vp,
                                                    int kt, __bf16* dst, int tid) {
#if HAS_ASYNC_LDS
  const uint4* src = (const uint4*)(vp + (size_t)kt * Dd);
  uint4* d = (uint4*)dst;
  for (int i = tid; i < (64 * Dd) / 8; i += 128)
    __builtin_amdgcn_global_load_async_to_lds_b128((gv4i_ptr)(src + i),
                                                   (lv4i_ptr)(d + i), 0, 0);
#else
  (void)vp; (void)kt; (void)dst; (void)tid;
#endif
}

// Make the V slice [kt, kt+64) resident in `dst`. Async path: wait for the
// oldest 32 per-wave async ops (tile kt); the 32 prefetch ops for tile kt+64
// may remain in flight (async loads complete in order).
static __device__ __forceinline__ void v_tile_wait(const __bf16* __restrict__ vp,
                                                   int kt, __bf16* dst, int tid,
                                                   bool more_inflight) {
#if HAS_ASYNC_LDS
  (void)vp; (void)kt; (void)dst; (void)tid;
  if (more_inflight) __builtin_amdgcn_s_wait_asynccnt(32);
  else               __builtin_amdgcn_s_wait_asynccnt(0);
#else
  (void)more_inflight;
  const uint4* src = (const uint4*)(vp + (size_t)kt * Dd);
  uint4* d = (uint4*)dst;
  for (int i = tid; i < (64 * Dd) / 8; i += 128) d[i] = src[i];
#endif
}

// ---------------------------------------------------------------------------
// Generic GEMM: C_bf16(M x 512) = A_f32(M x 512) @ W_f32(512 x 512)
// block = 128 threads (4 waves); wave -> one 16x16 tile; blockIdx.x covers 64
// columns, blockIdx.y a 16-row tile. A tile staged in LDS as bf16.
// ---------------------------------------------------------------------------
__global__ void gemm_f32_to_bf16(const float* __restrict__ A,
                                 const float* __restrict__ W,
                                 __bf16* __restrict__ C,
                                 const int* __restrict__ flag) {
  if (flag && *flag) return;
  __shared__ __bf16 As[16 * Dd];
  const int tid = threadIdx.x;
  const int row0 = blockIdx.y * 16;
  for (int i = tid; i < 16 * Dd; i += 128) {
    int r = i >> 9, c = i & 511;
    As[i] = (__bf16)A[(size_t)(row0 + r) * Dd + c];
  }
  __syncthreads();

  const int wave = tid >> 5, lane = tid & 31;
  const int lm = lane & 15, lh = lane >> 4;
  const int col0 = blockIdx.x * 64 + wave * 16;

  v8f acc = {};
  for (int k0 = 0; k0 < Dd; k0 += 32) {
    v16bf a, b;
#pragma unroll
    for (int e = 0; e < 16; ++e) {
      const int ka = lh * 16 + e;                          // A 16x32 bf16 layout
      a[e] = As[lm * Dd + k0 + ka];
      const int kb = lh * 16 + e;                          // B 32x16 bf16 layout
      b[e] = (__bf16)W[(size_t)(k0 + kb) * Dd + col0 + lm];
    }
    acc = wmma_bf16(a, b, acc);
  }
#pragma unroll
  for (int r = 0; r < 8; ++r) {
    const int row = row0 + r + 8 * lh;                     // C/D tile layout
    C[(size_t)row * Dd + col0 + lm] = (__bf16)acc[r];
  }
}

// ---------------------------------------------------------------------------
// Flash-attention step:  zout = softmax(beta * qb @ kpT) @ vp
// block = 128 threads (4 waves), one 16-row tile per block (grid.x = B/16).
// Per 64-wide K-tile: async prefetch of next V slice into the alternate LDS
// buffer, S tile via WMMA (DH=512 inner), online softmax, accumulator rescale,
// s_wait_asynccnt, P@V via WMMA (each wave owns 128 output columns).
// ---------------------------------------------------------------------------
__global__ void attn_step(const __bf16* __restrict__ qb,   // B x 512
                          const __bf16* __restrict__ kp,   // K x 512  (k_proj)
                          const __bf16* __restrict__ vp,   // K x 512  (v_proj)
                          const float*  __restrict__ zin,  // B x 512 or null
                          float*        __restrict__ zout, // B x 512
                          const float*  __restrict__ log_beta,
                          float* __restrict__ dpart,       // [grid] or null
                          float* __restrict__ npart,       // [grid] or null
                          const int* __restrict__ flag) {
  if (flag && *flag) return;
  extern __shared__ char smem[];
  __bf16* qs   = (__bf16*)(smem);                  // 16 x 512 bf16   (16 KB)
  __bf16* vs0  = (__bf16*)(smem + 16384);          // 64 x 512 bf16   (64 KB)
  __bf16* vs1  = (__bf16*)(smem + 81920);          // 64 x 512 bf16   (64 KB)
  float*  ss   = (float*) (smem + 147456);         // 16 x 64  f32    ( 4 KB)
  __bf16* ps   = (__bf16*)(smem + 151552);         // 16 x 64  bf16   ( 2 KB)
  float*  mrow = (float*) (smem + 153600);         // 16
  float*  lrow = mrow + 16;                        // 16
  float*  crow = lrow + 16;                        // 16

  const int tid = threadIdx.x;
  const int wave = tid >> 5, lane = tid & 31;
  const int lm = lane & 15, lh = lane >> 4;
  const int row0 = blockIdx.x * 16;
  const float beta = __expf(log_beta[0]);

  for (int i = tid; i < 16 * Dd; i += 128)
    qs[i] = qb[(size_t)row0 * Dd + i];
  if (tid < 16) { mrow[tid] = -3.0e38f; lrow[tid] = 0.0f; }

  v8f acc[8];
#pragma unroll
  for (int t = 0; t < 8; ++t) acc[t] = (v8f){};

  v_tile_issue(vp, 0, vs0, tid);   // kick off tile 0 while we set up
  __syncthreads();

  int ibuf = 0;
  for (int kt = 0; kt < Kpat; kt += 64, ibuf ^= 1) {
    __bf16* vcur = ibuf ? vs1 : vs0;
    __bf16* vnxt = ibuf ? vs0 : vs1;
    const bool more = (kt + 64) < Kpat;
    if (more) {
      v_tile_issue(vp, kt + 64, vnxt, tid);        // overlap with S phase
      const char* kpn = (const char*)(kp + (size_t)(kt + 64) * Dd);
      __builtin_prefetch(kpn + tid * 512, 0, 1);   // global_prefetch_b8
      __builtin_prefetch(kpn + tid * 512 + 256, 0, 1);
    }

    // ---- S = beta * q(16x512) @ kT(512x64): wave w -> columns [w*16, w*16+16)
    v8f sacc = {};
    for (int k0 = 0; k0 < Dd; k0 += 32) {
      v16bf a, b;
#pragma unroll
      for (int e = 0; e < 16; ++e) {
        const int ka = lh * 16 + e;
        a[e] = qs[lm * Dd + k0 + ka];
        const int kb = lh * 16 + e;
        // B element (kb, lm) of kT chunk == kp[kt + wave*16 + lm][k0 + kb]
        b[e] = kp[(size_t)(kt + wave * 16 + lm) * Dd + (k0 + kb)];
      }
      sacc = wmma_bf16(a, b, sacc);
    }
#pragma unroll
    for (int r = 0; r < 8; ++r)
      ss[(r + 8 * lh) * 64 + wave * 16 + lm] = beta * sacc[r];
    __syncthreads();

    // ---- online softmax (one thread per row, 64 cols)
    if (tid < 16) {
      const int r = tid;
      const float m_old = mrow[r];
      float mx = m_old;
      for (int c = 0; c < 64; ++c) mx = fmaxf(mx, ss[r * 64 + c]);
      float sum = 0.0f;
      for (int c = 0; c < 64; ++c) {
        const float p = __expf(ss[r * 64 + c] - mx);
        ps[r * 64 + c] = (__bf16)p;
        sum += p;
      }
      const float cf = __expf(m_old - mx);
      lrow[r] = lrow[r] * cf + sum;
      mrow[r] = mx;
      crow[r] = cf;
    }
    __syncthreads();

    // ---- rescale accumulators by exp(m_old - m_new)
#pragma unroll
    for (int t = 0; t < 8; ++t)
#pragma unroll
      for (int r = 0; r < 8; ++r) acc[t][r] *= crow[r + 8 * lh];

    // ---- ensure V[kt:kt+64, :] resident (async wait / sync fallback copy)
    v_tile_wait(vp, kt, vcur, tid, more);
    __syncthreads();

    // ---- P(16x64) @ V(64x16-col tiles): wave owns cols [wave*128, +128)
    v16bf pa[2];
#pragma unroll
    for (int ch = 0; ch < 2; ++ch)
#pragma unroll
      for (int e = 0; e < 16; ++e) {
        const int ka = lh * 16 + e;
        pa[ch][e] = ps[lm * 64 + ch * 32 + ka];
      }
#pragma unroll
    for (int t = 0; t < 8; ++t) {
      const int col0 = wave * 128 + t * 16;
#pragma unroll
      for (int ch = 0; ch < 2; ++ch) {
        v16bf b;
#pragma unroll
        for (int e = 0; e < 16; ++e) {
          const int kb = lh * 16 + e;
          b[e] = vcur[(ch * 32 + kb) * Dd + col0 + lm];
        }
        acc[t] = wmma_bf16(pa[ch], b, acc[t]);
      }
    }
    __syncthreads();
  }

  // ---- epilogue: normalize, optional delta-norm partials, store
  float d2 = 0.0f, n2 = 0.0f;
#pragma unroll
  for (int t = 0; t < 8; ++t) {
    const int col0 = wave * 128 + t * 16;
#pragma unroll
    for (int r = 0; r < 8; ++r) {
      const int row = row0 + r + 8 * lh;
      const float val = acc[t][r] / lrow[r + 8 * lh];
      const size_t idx = (size_t)row * Dd + col0 + lm;
      if (zin) {
        const float dv = val - zin[idx];
        d2 += dv * dv;
        n2 += val * val;
      }
      zout[idx] = val;
    }
  }
  if (dpart) {
    // deterministic in-block reduction (fixed order), ss is free now
    ss[tid] = d2;
    ss[128 + tid] = n2;
    __syncthreads();
    if (tid == 0) {
      float sd = 0.0f, sn = 0.0f;
      for (int i = 0; i < 128; ++i) { sd += ss[i]; sn += ss[128 + i]; }
      dpart[blockIdx.x] = sd;
      npart[blockIdx.x] = sn;
    }
  }
}

__global__ void init_flag(int* flag) { *flag = 0; }

__global__ void check_converged(const float* __restrict__ dpart,
                                const float* __restrict__ npart,
                                int* flag) {
  if (*flag) return;
  float sd = 0.0f, sn = 0.0f;
  for (int i = 0; i < Bsz / 16; ++i) { sd += dpart[i]; sn += npart[i]; }
  const float rel = sqrtf(sd) / (sqrtf(sn) + 1e-8f);
  if (rel <= TOLf) *flag = 1;
}

// ---------------------------------------------------------------------------
// Gate: one wave32 per row; lane == hidden unit (GH == 32 == wave width).
// ---------------------------------------------------------------------------
__global__ void gate_kernel(const float* __restrict__ shallow,
                            const float* __restrict__ deep,
                            const float* __restrict__ g1w,  // (1025, 32)
                            const float* __restrict__ g1b,  // (32)
                            const float* __restrict__ g2w,  // (32)
                            const float* __restrict__ g2b,  // (1)
                            float* __restrict__ out) {
  const int row = blockIdx.x;
  const int lane = threadIdx.x;
  const float* sr = shallow + (size_t)row * Dd;
  const float* dr = deep + (size_t)row * Dd;

  float loc = 0.0f;
  for (int c = lane; c < Dd; c += 32) {
    const float d = sr[c] - dr[c];
    loc += d * d;
  }
#pragma unroll
  for (int off = 16; off > 0; off >>= 1) loc += __shfl_down(loc, off, 32);
  const float div = sqrtf(__shfl(loc, 0, 32));

  float h = g1b[lane];
  for (int i = 0; i < Dd; ++i) h += sr[i] * g1w[i * GHd + lane];
  for (int i = 0; i < Dd; ++i) h += dr[i] * g1w[(Dd + i) * GHd + lane];
  h += div * g1w[2 * Dd * GHd + lane];
  h = 0.5f * h * (1.0f + erff(h * 0.70710678118654752f));  // exact GELU

  float s = h * g2w[lane];
#pragma unroll
  for (int off = 16; off > 0; off >>= 1) s += __shfl_down(s, off, 32);
  const float alpha = 1.0f / (1.0f + expf(-(__shfl(s, 0, 32) + g2b[0])));

  for (int c = lane; c < Dd; c += 32)
    out[(size_t)row * Dd + c] = alpha * sr[c] + (1.0f - alpha) * dr[c];
}

// ---------------------------------------------------------------------------
extern "C" void kernel_launch(void* const* d_in, const int* in_sizes, int n_in,
                              void* d_out, int out_size, void* d_ws,
                              size_t ws_size, hipStream_t stream) {
  (void)in_sizes; (void)n_in; (void)out_size; (void)ws_size;
  const float* query    = (const float*)d_in[0];
  const float* patterns = (const float*)d_in[1];
  const float* Wq       = (const float*)d_in[2];
  const float* Wk       = (const float*)d_in[3];
  const float* Wv       = (const float*)d_in[4];
  const float* log_beta = (const float*)d_in[5];
  const float* g1w      = (const float*)d_in[6];
  const float* g1b      = (const float*)d_in[7];
  const float* g2w      = (const float*)d_in[8];
  const float* g2b      = (const float*)d_in[9];

  const size_t BD = (size_t)Bsz * Dd;
  float* out_output  = (float*)d_out;
  float* out_shallow = out_output + BD;
  float* out_deep    = out_shallow + BD;   // doubles as the z buffer

  // workspace carve-up
  char* w = (char*)d_ws;
  __bf16* kp = (__bf16*)w;                  w += (size_t)Kpat * Dd * 2;  // 16 MB
  __bf16* vp = (__bf16*)w;                  w += (size_t)Kpat * Dd * 2;  // 16 MB
  __bf16* qb = (__bf16*)w;                  w += (size_t)Bsz * Dd * 2;   //  4 MB
  float* dpart = (float*)w;                 w += (Bsz / 16) * 4;
  float* npart = (float*)w;                 w += (Bsz / 16) * 4;
  int* flag = (int*)w;

  const int SMEM = 153792;  // qs + vs0 + vs1 + ss + ps + m/l/c rows
  (void)hipFuncSetAttribute((const void*)attn_step,
                            hipFuncAttributeMaxDynamicSharedMemorySize, SMEM);

  hipLaunchKernelGGL(init_flag, dim3(1), dim3(1), 0, stream, flag);

  // projections: k_proj = patterns@Wk, v_proj = patterns@Wv (bf16)
  hipLaunchKernelGGL(gemm_f32_to_bf16, dim3(Dd / 64, Kpat / 16), dim3(128), 0,
                     stream, patterns, Wk, kp, (const int*)nullptr);
  hipLaunchKernelGGL(gemm_f32_to_bf16, dim3(Dd / 64, Kpat / 16), dim3(128), 0,
                     stream, patterns, Wv, vp, (const int*)nullptr);

  // shallow pass: q = query@Wq; out_shallow = attn(q)
  hipLaunchKernelGGL(gemm_f32_to_bf16, dim3(Dd / 64, Bsz / 16), dim3(128), 0,
                     stream, query, Wq, qb, (const int*)nullptr);
  hipLaunchKernelGGL(attn_step, dim3(Bsz / 16), dim3(128), SMEM, stream, qb, kp,
                     vp, (const float*)nullptr, out_shallow, log_beta,
                     (float*)nullptr, (float*)nullptr, (const int*)nullptr);

  // z0 = out_shallow (z lives in the out_deep slot)
  (void)hipMemcpyAsync(out_deep, out_shallow, BD * sizeof(float),
                       hipMemcpyDeviceToDevice, stream);

  // fixed-point iterations with device-side early-exit flag
  for (int it = 0; it < 30; ++it) {
    hipLaunchKernelGGL(gemm_f32_to_bf16, dim3(Dd / 64, Bsz / 16), dim3(128), 0,
                       stream, out_deep, Wq, qb, (const int*)flag);
    hipLaunchKernelGGL(attn_step, dim3(Bsz / 16), dim3(128), SMEM, stream, qb,
                       kp, vp, out_deep, out_deep, log_beta, dpart, npart,
                       (const int*)flag);
    hipLaunchKernelGGL(check_converged, dim3(1), dim3(1), 0, stream, dpart,
                       npart, flag);
  }

  // gate + blend
  hipLaunchKernelGGL(gate_kernel, dim3(Bsz), dim3(32), 0, stream, out_shallow,
                     out_deep, g1w, g1b, g2w, g2b, out_output);
}
